// fA_P4DenseNet_78572131713149
// MI455X (gfx1250) — matrix operand order repaired and checked by reading, not compiled, not run on hardware
//
#include <hip/hip_runtime.h>
#include <hip/hip_bf16.h>

// ---------------------------------------------------------------------------
// P4-equivariant DenseNet forward for MI455X (gfx1250).
// Heavy group convolutions: implicit GEMM on v_wmma_f32_16x16x32_f16 with
// channels-last f16 activations so both A and B fragments are pure b128 loads.
// Elementwise / reduction stages: plain VALU kernels (HBM-bound anyway).
// ---------------------------------------------------------------------------

#define NGROT 4

typedef __attribute__((ext_vector_type(16))) _Float16 v16h;
typedef __attribute__((ext_vector_type(8)))  _Float16 v8h;
typedef __attribute__((ext_vector_type(8)))  float    v8f;

static __device__ __forceinline__ float dsigmoid(float x) {
  return 1.0f / (1.0f + __expf(-x));
}

// ------------------------- BN statistics (deterministic) -------------------
__global__ void k_stats_partial(const float* __restrict__ x,
                                float* __restrict__ ps, float* __restrict__ pq,
                                int C, long GHW, long per, long CHUNK, int nchunks) {
  int c = blockIdx.x;
  long start = (long)blockIdx.y * CHUNK;
  long end = start + CHUNK; if (end > per) end = per;
  float s = 0.f, q = 0.f;
  for (long e = start + threadIdx.x; e < end; e += blockDim.x) {
    long b = e / GHW; long rem = e - b * GHW;
    float v = x[(b * C + c) * GHW + rem];
    s += v; q += v * v;
  }
  __shared__ float rs[256], rq[256];
  rs[threadIdx.x] = s; rq[threadIdx.x] = q;
  __syncthreads();
  for (int o = 128; o > 0; o >>= 1) {
    if ((int)threadIdx.x < o) { rs[threadIdx.x] += rs[threadIdx.x + o]; rq[threadIdx.x] += rq[threadIdx.x + o]; }
    __syncthreads();
  }
  if (threadIdx.x == 0) {
    ps[(long)c * nchunks + blockIdx.y] = rs[0];
    pq[(long)c * nchunks + blockIdx.y] = rq[0];
  }
}

__global__ void k_stats_fin(const float* __restrict__ ps, const float* __restrict__ pq,
                            float* __restrict__ sums, float* __restrict__ sqs,
                            int C, int nchunks) {
  int c = blockIdx.x * blockDim.x + threadIdx.x;
  if (c >= C) return;
  float s = 0.f, q = 0.f;
  for (int j = 0; j < nchunks; ++j) { s += ps[(long)c * nchunks + j]; q += pq[(long)c * nchunks + j]; }
  sums[c] = s; sqs[c] = q;
}

__global__ void k_bn_relu(const float* __restrict__ x, float* __restrict__ y,
                          const float* __restrict__ sums, const float* __restrict__ sqs,
                          const float* __restrict__ gamma, const float* __restrict__ beta,
                          int C, long GHW, long total, float invcnt) {
  long i = (long)blockIdx.x * blockDim.x + threadIdx.x;
  if (i >= total) return;
  int c = (int)((i / GHW) % C);
  float m = sums[c] * invcnt;
  float v = sqs[c] * invcnt - m * m;
  float t = (x[i] - m) * rsqrtf(v + 2e-5f) * gamma[c] + beta[c];
  y[i] = t > 0.f ? t : 0.f;
}

// ------------------------- channel attention -------------------------------
__global__ void k_spatial_mean(const float* __restrict__ x, float* __restrict__ z, long HW) {
  long base = (long)blockIdx.x * HW;
  float s = 0.f;
  for (long i = threadIdx.x; i < HW; i += blockDim.x) s += x[base + i];
  __shared__ float rs[256];
  rs[threadIdx.x] = s; __syncthreads();
  for (int o = 128; o > 0; o >>= 1) {
    if ((int)threadIdx.x < o) rs[threadIdx.x] += rs[threadIdx.x + o];
    __syncthreads();
  }
  if (threadIdx.x == 0) z[blockIdx.x] = rs[0] / (float)HW;
}

__global__ void k_mlp_gate(const float* __restrict__ z, const float* __restrict__ w1,
                           const float* __restrict__ w2, float* __restrict__ a,
                           int CG, int HID) {
  int b = blockIdx.x;
  __shared__ float sm[8];
  if ((int)threadIdx.x < HID) {
    float h = 0.f;
    for (int cg = 0; cg < CG; ++cg) h += z[(long)b * CG + cg] * w1[cg * HID + threadIdx.x];
    sm[threadIdx.x] = h > 0.f ? h : 0.f;
  }
  __syncthreads();
  for (int cg = threadIdx.x; cg < CG; cg += blockDim.x) {
    float s = 0.f;
    for (int j = 0; j < HID; ++j) s += sm[j] * w2[j * CG + cg];
    a[(long)b * CG + cg] = dsigmoid(s);
  }
}

// ------------------------- spatial attention -------------------------------
// f[b][d*G+g][hw] = mean / max over channel c of x[b][c*G+g][hw]
__global__ void k_meanmax(const float* __restrict__ x, float* __restrict__ f,
                          int B, int C, int G, long HW) {
  long tot = (long)B * G * HW;
  long i = (long)blockIdx.x * blockDim.x + threadIdx.x;
  if (i >= tot) return;
  long hw = i % HW; long t = i / HW;
  int g = (int)(t % G); long b = t / G;
  float s = 0.f, mx = -3.402823e38f;
  for (int c = 0; c < C; ++c) {
    float v = x[(((long)b * C + c) * G + g) * HW + hw];
    s += v; mx = fmaxf(mx, v);
  }
  f[((long)b * 2 * G + g) * HW + hw] = s / (float)C;
  f[((long)b * 2 * G + G + g) * HW + hw] = mx;
}

// wk[gout][d][g][i][j] = roll(rot90(w, gout), gout, axis=G)
__global__ void k_rot_sa(const float* __restrict__ w, float* __restrict__ wk,
                         int G, int ks) {
  int tot = 4 * 2 * G * ks * ks;
  int i = blockIdx.x * blockDim.x + threadIdx.x;
  if (i >= tot) return;
  int kj = i % ks; int t = i / ks;
  int ki = t % ks; t /= ks;
  int g = t % G; t /= G;
  int d = t % 2; int gout = t / 2;
  int gsrc = (g - gout) & 3;
  int si = ki, sj = kj;
  for (int r = 0; r < gout; ++r) { int ni = sj; int nj = ks - 1 - si; si = ni; sj = nj; }
  wk[i] = w[((d * G + gsrc) * ks + si) * ks + sj];
}

__global__ void k_sa_conv(const float* __restrict__ f, const float* __restrict__ wk,
                          float* __restrict__ sa, int B, int CIN, int G,
                          int H, int W, int ks, int dil) {
  long tot = (long)B * G * H * W;
  long i = (long)blockIdx.x * blockDim.x + threadIdx.x;
  if (i >= tot) return;
  int w0 = (int)(i % W); long t = i / W;
  int h0 = (int)(t % H); t /= H;
  int g = (int)(t % G); long b = t / G;
  int half = ks / 2;
  const float* wg = wk + (long)g * CIN * ks * ks;
  float acc = 0.f;
  for (int c2 = 0; c2 < CIN; ++c2)
    for (int ki = 0; ki < ks; ++ki) {
      int hh = h0 + dil * (ki - half);
      if (hh < 0 || hh >= H) continue;
      for (int kj = 0; kj < ks; ++kj) {
        int ww = w0 + dil * (kj - half);
        if (ww < 0 || ww >= W) continue;
        acc += f[((long)b * CIN + c2) * H * W + (long)hh * W + ww] * wg[(c2 * ks + ki) * ks + kj];
      }
    }
  sa[((long)b * G + g) * H * W + (long)h0 * W + w0] = dsigmoid(acc);
}

// gate for the lifting conv (stays f32, tiny)
__global__ void k_gate(const float* __restrict__ in, float* __restrict__ out,
                       const float* __restrict__ ca, const float* __restrict__ sa,
                       int B, int CG, int G, long HW) {
  long tot = (long)B * CG * HW;
  long i = (long)blockIdx.x * blockDim.x + threadIdx.x;
  if (i >= tot) return;
  long hw = i % HW; long t = i / HW;
  int cg = (int)(t % CG); long b = t / CG;
  int g = cg % G;
  out[i] = in[i] * ca[(long)b * CG + cg] * sa[((long)b * G + g) * HW + hw];
}

// gate + transpose to channels-last f16: xal[b][h][w][cg] = y*ca*sa
__global__ void k_gate_f16(const float* __restrict__ y, unsigned short* __restrict__ xal,
                           const float* __restrict__ ca, const float* __restrict__ sa,
                           int B, int CG, int G, long HW) {
  long tot = (long)B * HW * CG;
  long i = (long)blockIdx.x * blockDim.x + threadIdx.x;
  if (i >= tot) return;
  int cg = (int)(i % CG); long t = i / CG;
  long hw = t % HW; long b = t / HW;
  int g = cg % G;
  float v = y[((long)b * CG + cg) * HW + hw] * ca[(long)b * CG + cg] *
            sa[((long)b * G + g) * HW + hw];
  reinterpret_cast<_Float16*>(xal)[i] = (_Float16)v;
}

// ------------------------- weight prep (rot90 + group roll, f32 -> f16) ----
// w: [NOUT][C][G][KH][KH] -> wrot: [4][Mpad][K], K-order k = rr*CG + cg
// (rr = spatial tap, cg-minor, matching channels-last activations).
__global__ void k_rot_wgg(const float* __restrict__ w, unsigned short* __restrict__ wrot,
                          int NOUT, int C, int G, int KH, int Mpad) {
  int KHW = KH * KH;
  int CG = C * G;
  long K = (long)CG * KHW;
  long tot = 4L * Mpad * K;
  long i = (long)blockIdx.x * blockDim.x + threadIdx.x;
  if (i >= tot) return;
  long k = i % K; long t = i / K;
  int m = (int)(t % Mpad); int r = (int)(t / Mpad);
  _Float16 val = (_Float16)0.f;
  if (m < NOUT) {
    int rr = (int)(k / CG);
    int cg = (int)(k - (long)rr * CG);
    int kh = rr / KH, kw = rr - (rr / KH) * KH;
    int c = cg / G, g = cg - (cg / G) * G;
    int gsrc = (g - r) & 3;
    int si = kh, sj = kw;
    for (int q = 0; q < r; ++q) { int ni = sj; int nj = KH - 1 - si; si = ni; sj = nj; }
    val = (_Float16)w[((((long)m * C + c) * G + gsrc) * KH + si) * KH + sj];
  }
  reinterpret_cast<_Float16*>(wrot)[i] = val;
}

// ------------------------- WMMA implicit-GEMM group conv -------------------
// D[m, n] = sum_k wrot[r][m][k] * xal[b(n)][oh(n)+kh(k)][ow(n)+kw(k)][cg(k)]
// CG is always a multiple of 16 here, so every lane's two 8-wide K runs (k = 8h+s, 16+8h+s)
// are contiguous f16 in memory: A and B fragments are each two b128 loads.
// Grid: (ceil(B*OH*OW/64), Mpad/16, 4).  Block: 128 (4 waves, 1 tile each).
template <int KHW>
__global__ void __launch_bounds__(128)
k_wmma_conv(const unsigned short* __restrict__ xal, const unsigned short* __restrict__ wrot,
            float* __restrict__ out, int B, int CG, int H, int W,
            int NOUT, int Mpad, int OH, int OW) {
  const int KW = (KHW == 9) ? 3 : 1;
  const int K = CG * KHW;
  const long Ntot = (long)B * OH * OW;

  const int lane = threadIdx.x & 31;
  const int wv = threadIdx.x >> 5;
  const int r = blockIdx.z;
  const int l15 = lane & 15;
  const int khalf = lane >> 4;

  long n0 = (long)blockIdx.x * 64 + wv * 16;
  long ncol = n0 + l15;
  bool nvalid = ncol < Ntot;
  long nc = nvalid ? ncol : 0;
  int b = (int)(nc / ((long)OH * OW));
  int rem = (int)(nc - (long)b * OH * OW);
  int oh = rem / OW, ow = rem - (rem / OW) * OW;

  const int mrow = blockIdx.y * 16 + l15;
  const _Float16* wr = reinterpret_cast<const _Float16*>(wrot) +
                       ((long)r * Mpad + mrow) * (long)K;
  const _Float16* xb = reinterpret_cast<const _Float16*>(xal) +
                       (long)b * H * W * CG;

  v8f acc = {};
#pragma unroll
  for (int rr = 0; rr < KHW; ++rr) {
    const int kh = rr / KW, kw = rr - (rr / KW) * KW;
    const _Float16* pB = xb + ((long)(oh + kh) * W + (ow + kw)) * CG;
    const _Float16* pA = wr + (long)rr * CG;
    for (int cgb = 0; cgb < CG; cgb += 32) {
      __builtin_prefetch(pA + cgb + 128, 0, 1);
      v16h Af, Bf;
      *(v8h*)&Af       = *(const v8h*)(pA + cgb + khalf * 8);
      *((v8h*)&Af + 1) = *(const v8h*)(pA + cgb + khalf * 8 + 16);
      *(v8h*)&Bf       = *(const v8h*)(pB + cgb + khalf * 8);
      *((v8h*)&Bf + 1) = *(const v8h*)(pB + cgb + khalf * 8 + 16);
      acc = __builtin_amdgcn_wmma_f32_16x16x32_f16(false, Af, false, Bf,
                                                   (short)0, acc, false, false);
    }
  }
  // ---- store: lane L -> col n0+L%16, rows (L/16)*8 + i ----
  if (nvalid) {
    int mbase = blockIdx.y * 16 + khalf * 8;
#pragma unroll
    for (int i = 0; i < 8; ++i) {
      int m = mbase + i;
      if (m < NOUT)
        out[(((long)b * NOUT + m) * NGROT + r) * ((long)OH * OW) + (long)oh * OW + ow] = acc[i];
    }
  }
}

// ------------------------- lifting conv (c1) -------------------------------
__global__ void k_rot_c1(const float* __restrict__ w, float* __restrict__ wr) {
  int tot = 4 * 24 * 3 * 9;
  int i = blockIdx.x * blockDim.x + threadIdx.x;
  if (i >= tot) return;
  int kj = i % 3; int t = i / 3;
  int ki = t % 3; t /= 3;
  int c = t % 3; t /= 3;
  int m = t % 24; int r = t / 24;
  int si = ki, sj = kj;
  for (int q = 0; q < r; ++q) { int ni = sj; int nj = 2 - si; si = ni; sj = nj; }
  wr[i] = w[((m * 3 + c) * 3 + si) * 3 + sj];
}

__global__ void k_conv_rng(const float* __restrict__ xa, const float* __restrict__ wr,
                           float* __restrict__ out, int B, int H, int OH) {
  int W = H, OW = OH;
  long tot = (long)B * 24 * NGROT * OH * OW;
  long i = (long)blockIdx.x * blockDim.x + threadIdx.x;
  if (i >= tot) return;
  int ow = (int)(i % OW); long t = i / OW;
  int oh = (int)(t % OH); t /= OH;
  int r = (int)(t % NGROT); t /= NGROT;
  int m = (int)(t % 24); long b = t / 24;
  float acc = 0.f;
  for (int c = 0; c < 3; ++c)
    for (int ki = 0; ki < 3; ++ki)
      for (int kj = 0; kj < 3; ++kj)
        acc += xa[((long)b * 3 + c) * H * W + (long)(oh + ki) * W + (ow + kj)] *
               wr[(((r * 24 + m) * 3 + c) * 3 + ki) * 3 + kj];
  out[i] = acc;
}

// ------------------------- concat / pool / final ---------------------------
__global__ void k_concat(const float* __restrict__ h, const float* __restrict__ x,
                         float* __restrict__ o, int B, int Cin, int G, int H, int W) {
  int OH = H - 2, OW = W - 2, Cn = Cin + 24;
  long tot = (long)B * Cn * G * OH * OW;
  long i = (long)blockIdx.x * blockDim.x + threadIdx.x;
  if (i >= tot) return;
  int ow = (int)(i % OW); long t = i / OW;
  int oh = (int)(t % OH); t /= OH;
  int g = (int)(t % G); t /= G;
  int c = (int)(t % Cn); long b = t / Cn;
  float v;
  if (c < 24)
    v = h[(((b * 24 + c) * G + g) * (long)OH + oh) * OW + ow];
  else
    v = x[(((b * Cin + (c - 24)) * G + g) * (long)H + (oh + 1)) * W + (ow + 1)];
  o[i] = v;
}

__global__ void k_avgpool2(const float* __restrict__ h, float* __restrict__ o,
                           int B, int C, int G, int H, int W) {
  int OH = H / 2, OW = W / 2;
  long tot = (long)B * C * G * OH * OW;
  long i = (long)blockIdx.x * blockDim.x + threadIdx.x;
  if (i >= tot) return;
  int ow = (int)(i % OW); long t = i / OW;
  int oh = (int)(t % OH); long ch = t / OH;
  const float* s = h + ch * (long)H * W;
  float v = s[(long)(2 * oh) * W + 2 * ow] + s[(long)(2 * oh) * W + 2 * ow + 1] +
            s[(long)(2 * oh + 1) * W + 2 * ow] + s[(long)(2 * oh + 1) * W + 2 * ow + 1];
  o[i] = 0.25f * v;
}

__global__ void k_final(const float* __restrict__ h, float* __restrict__ o, int B) {
  int i = blockIdx.x * blockDim.x + threadIdx.x;
  if (i >= B * 2) return;
  float s = 0.f;
  for (int t = 0; t < 16; ++t) s += h[(long)i * 16 + t];
  o[i] = s * (1.f / 16.f);
}

// ===========================================================================
extern "C" void kernel_launch(void* const* d_in, const int* in_sizes, int n_in,
                              void* d_out, int out_size, void* d_ws, size_t ws_size,
                              hipStream_t stream) {
  (void)in_sizes; (void)n_in; (void)out_size; (void)ws_size;
  const int B = 32;

  // ---- inputs (setup_inputs() flattening order) ----
  int pi = 0;
  auto IN = [&]() { return (const float*)d_in[pi++]; };
  const float* x = IN();
  const float* c1w = IN(); const float* c1ca1 = IN(); const float* c1ca2 = IN(); const float* c1sa = IN();
  struct BP { const float *gamma, *beta, *w, *ca1, *ca2, *sa; };
  BP blk[5], trn[4];
  for (int i = 0; i < 5; ++i) { blk[i].gamma = IN(); blk[i].beta = IN(); blk[i].w = IN(); blk[i].ca1 = IN(); blk[i].ca2 = IN(); blk[i].sa = IN(); }
  for (int i = 0; i < 4; ++i) { trn[i].gamma = IN(); trn[i].beta = IN(); trn[i].w = IN(); trn[i].ca1 = IN(); trn[i].ca2 = IN(); trn[i].sa = IN(); }
  const float* bng = IN(); const float* bnb = IN();
  const float* cow = IN(); const float* coca1 = IN(); const float* coca2 = IN(); const float* cosa = IN();

  // ---- workspace bump allocator ----
  char* wsb = (char*)d_ws;
  size_t off = 0;
  auto alloc = [&](size_t bytes) -> void* {
    void* r = wsb + off;
    off = (off + bytes + 255) & ~(size_t)255;
    return r;
  };
  const size_t FEAT = 52002816;  // [32,48,4,92,92] (largest live feature map)
  float* bufA = (float*)alloc(FEAT * 4);
  float* bufB = (float*)alloc(FEAT * 4);
  float* bufH = (float*)alloc(FEAT * 4);
  float* bufY = (float*)alloc(FEAT * 4);
  unsigned short* xal = (unsigned short*)alloc(FEAT * 2);  // gated f16 channels-last
  float* bufSA = (float*)alloc((size_t)1200000 * 4);
  float* bufF = (float*)alloc((size_t)2400000 * 4);
  float* bufZ = (float*)alloc((size_t)B * 576 * 4);
  float* bufCA = (float*)alloc((size_t)B * 576 * 4);
  float* psum = (float*)alloc((size_t)144 * 128 * 4);
  float* pqsum = (float*)alloc((size_t)144 * 128 * 4);
  float* sums = (float*)alloc(256 * 4);
  float* sqs = (float*)alloc(256 * 4);
  unsigned short* wrotbuf = (unsigned short*)alloc((size_t)1200000 * 2);
  float* wkbuf = (float*)alloc(2048 * 4);
  float* wrotc1 = (float*)alloc(4096 * 4);

  auto g1 = [](long n, int t) { return dim3((unsigned)((n + t - 1) / t)); };

  // ---- bn3d_relu: cur -> bufY ----
  auto run_bn = [&](const float* xin, const float* gamma, const float* beta, int C, int H) {
    long HW = (long)H * H;
    long GHW = NGROT * HW;
    long per = (long)B * GHW;
    const long CHUNK = 256 * 64;
    int nch = (int)((per + CHUNK - 1) / CHUNK);
    k_stats_partial<<<dim3(C, nch), 256, 0, stream>>>(xin, psum, pqsum, C, GHW, per, CHUNK, nch);
    k_stats_fin<<<g1(C, 64), 64, 0, stream>>>(psum, pqsum, sums, sqs, C, nch);
    long total = (long)B * C * GHW;
    k_bn_relu<<<g1(total, 256), 256, 0, stream>>>(xin, bufY, sums, sqs, gamma, beta,
                                                  C, GHW, total, 1.0f / (float)per);
  };

  // ---- att_conv_gg: gates bufY -> xal (f16, channels-last), WMMA conv -----
  auto run_attconv = [&](BP P, int C, int H, int NOUT, int KH, int sa_ks, int sa_dil, float* hout) {
    int W = H; long HW = (long)H * W;
    int CG = C * NGROT;
    // channel attention
    k_spatial_mean<<<dim3((unsigned)(B * CG)), 256, 0, stream>>>(bufY, bufZ, HW);
    k_mlp_gate<<<dim3(B), 64, 0, stream>>>(bufZ, P.ca1, P.ca2, bufCA, CG, 2);
    // spatial attention (group-equivariant)
    k_meanmax<<<g1((long)B * NGROT * HW, 256), 256, 0, stream>>>(bufY, bufF, B, C, NGROT, HW);
    k_rot_sa<<<g1(4L * 2 * NGROT * sa_ks * sa_ks, 64), 64, 0, stream>>>(P.sa, wkbuf, NGROT, sa_ks);
    k_sa_conv<<<g1((long)B * NGROT * HW, 256), 256, 0, stream>>>(bufF, wkbuf, bufSA, B, 2 * NGROT,
                                                                 NGROT, H, W, sa_ks, sa_dil);
    // gate + transpose to channels-last f16
    k_gate_f16<<<g1((long)B * HW * CG, 256), 256, 0, stream>>>(bufY, xal, bufCA, bufSA,
                                                               B, CG, NGROT, HW);
    // rotated/rolled weights -> f16 GEMM A matrix (rr-major, cg-minor K)
    int KHW = KH * KH;
    long Kdim = (long)CG * KHW;
    int Mpad = (NOUT + 15) & ~15;
    k_rot_wgg<<<g1(4L * Mpad * Kdim, 256), 256, 0, stream>>>(P.w, wrotbuf, NOUT, C, NGROT, KH, Mpad);
    // implicit-GEMM conv via WMMA
    int OH = H - (KH - 1);
    long Ntot = (long)B * OH * OH;
    dim3 gw((unsigned)((Ntot + 63) / 64), Mpad / 16, NGROT);
    if (KH == 3)
      k_wmma_conv<9><<<gw, 128, 0, stream>>>(xal, wrotbuf, hout, B, CG, H, W, NOUT, Mpad, OH, OH);
    else
      k_wmma_conv<1><<<gw, 128, 0, stream>>>(xal, wrotbuf, hout, B, CG, H, W, NOUT, Mpad, OH, OH);
  };

  // ------------------ stage c1: lifting att-conv ------------------
  {
    int H = 96; long HW = (long)H * H;
    k_spatial_mean<<<dim3((unsigned)(B * 3)), 256, 0, stream>>>(x, bufZ, HW);
    k_mlp_gate<<<dim3(B), 64, 0, stream>>>(bufZ, c1ca1, c1ca2, bufCA, 3, 3);
    k_meanmax<<<g1((long)B * HW, 256), 256, 0, stream>>>(x, bufF, B, 3, 1, HW);
    k_sa_conv<<<g1((long)B * HW, 256), 256, 0, stream>>>(bufF, c1sa, bufSA, B, 2, 1, H, H, 7, 4);
    k_gate<<<g1((long)B * 3 * HW, 256), 256, 0, stream>>>(x, bufY, bufCA, bufSA, B, 3, 1, HW);
    k_rot_c1<<<g1(4 * 24 * 3 * 9, 64), 64, 0, stream>>>(c1w, wrotc1);
    long tot = (long)B * 24 * NGROT * 94 * 94;
    k_conv_rng<<<g1(tot, 256), 256, 0, stream>>>(bufY, wrotc1, bufA, B, 96, 94);
  }

  // ------------------ dense / transition pipeline ------------------
  const int dilB[5] = {4, 4, 3, 2, 1};
  const int dilT[4] = {4, 4, 3, 1};
  const int spks[5] = {7, 7, 7, 7, 5};
  int C = 24, H = 94;
  float* cur = bufA; float* nxt = bufB;

  for (int i = 0; i < 4; ++i) {
    // dense block
    run_bn(cur, blk[i].gamma, blk[i].beta, C, H);
    run_attconv(blk[i], C, H, 24, 3, spks[i], dilB[i], bufH);
    k_concat<<<g1((long)B * (C + 24) * NGROT * (H - 2) * (H - 2), 256), 256, 0, stream>>>(
        bufH, cur, nxt, B, C, NGROT, H, H);
    { float* t = cur; cur = nxt; nxt = t; }
    C += 24; H -= 2;
    // transition block
    run_bn(cur, trn[i].gamma, trn[i].beta, C, H);
    run_attconv(trn[i], C, H, C, 1, 7, dilT[i], bufH);
    k_avgpool2<<<g1((long)B * C * NGROT * (H / 2) * (H / 2), 256), 256, 0, stream>>>(
        bufH, nxt, B, C, NGROT, H, H);
    { float* t = cur; cur = nxt; nxt = t; }
    H /= 2;
  }

  // dense block 4
  run_bn(cur, blk[4].gamma, blk[4].beta, C, H);
  run_attconv(blk[4], C, H, 24, 3, spks[4], dilB[4], bufH);
  k_concat<<<g1((long)B * (C + 24) * NGROT * (H - 2) * (H - 2), 256), 256, 0, stream>>>(
      bufH, cur, nxt, B, C, NGROT, H, H);
  { float* t = cur; cur = nxt; nxt = t; }
  C += 24; H -= 2;  // C=144, H=2

  // output bn + att-conv (1x1) + global mean
  run_bn(cur, bng, bnb, C, H);
  BP co; co.gamma = nullptr; co.beta = nullptr; co.w = cow; co.ca1 = coca1; co.ca2 = coca2; co.sa = cosa;
  run_attconv(co, C, H, 2, 1, 3, 1, bufH);
  k_final<<<g1(B * 2, 64), 64, 0, stream>>>(bufH, (float*)d_out, B);
}
